// Attention_49091476193618
// MI455X (gfx1250) — compile-verified, matrix operand order checked by reading, not run on hardware
//
#include <hip/hip_runtime.h>
#include <hip/hip_bf16.h>

// ---------------------------------------------------------------------------
// MI455X (gfx1250) fused attention:
//   q = x@Wq+bq ; k = z@Wk+bk ; v = z@Wv+bv
//   y = softmax((q k^T)/sqrt(128)) v
// bf16 WMMA (v_wmma_f32_16x16x32_bf16) with f32 accumulation, flash-attention
// streaming over keys, S^T/y^T formulation so no cross-lane transposes needed.
// K/V^T tiles staged into LDS by the Tensor Data Mover (tensor_load_to_lds),
// synchronized with s_wait_tensorcnt.
// ---------------------------------------------------------------------------

typedef __bf16 bf16;
typedef __attribute__((ext_vector_type(16))) __bf16       v16bf;
typedef __attribute__((ext_vector_type(8)))  float        v8f;
typedef __attribute__((ext_vector_type(4))) unsigned int  u32x4;
typedef __attribute__((ext_vector_type(2))) unsigned int  u32x2;
typedef __attribute__((ext_vector_type(4)))  float        f32x4;
typedef __attribute__((ext_vector_type(8)))  int          i32x8;
typedef __attribute__((ext_vector_type(4)))  int          i32x4;

#define SEQ_L 4096
#define D_MODEL 1024
#define D_ATTN 128

union ABFrag { v16bf v; u32x4 u[2]; };   // one WMMA A/B operand (16 bf16/lane)
union Pack4  { u32x2 u; __bf16 h[4]; };
union Pack8  { u32x4 u; __bf16 h[8]; };

__device__ __forceinline__ v8f wmma_bf16(v16bf a, v16bf b, v8f c) {
  // D = A(16x32) * B(32x16) + C, f32 accumulate
  return __builtin_amdgcn_wmma_f32_16x16x32_bf16(false, a, false, b,
                                                 (short)0, c, false, false);
}

// ---------------------------------------------------------------------------
// Tensor Data Mover: issue a 1D/2D tile load (global -> LDS).
// D# packing per CDNA5 ISA ch.8. data_size units of 8 bytes (field value 3).
// tile_d1 == 0 -> 1D tile. All descriptor inputs are wave-uniform.
// This toolchain's builtin takes 6 args (g0, g1, g2, g3, extra, cpol).
// ---------------------------------------------------------------------------
__device__ __forceinline__ void tdm_load_2d(unsigned lds_addr, const void* gptr,
                                            unsigned tile_d0, unsigned tile_d1,
                                            unsigned long stride0_units) {
  unsigned long ga = (unsigned long)gptr;
  u32x4 g0;
  g0[0] = 1u;                                   // count=1, user mode, no gather
  g0[1] = lds_addr;                             // LDS byte address
  g0[2] = (unsigned)(ga & 0xffffffffu);         // global_addr[31:0]
  g0[3] = (unsigned)((ga >> 32) & 0x01ffffffu)  // global_addr[56:32]
          | (2u << 30);                         // type=2 ("image")
  i32x8 g1;
  const unsigned data_size = 3u;                // 8-byte elements
  const unsigned tensor_d0 = 0x7fffffffu;       // OOB never triggers
  const unsigned tensor_d1 = 0x7fffffffu;
  g1[0] = (int)(data_size << 16);               // wg_mask=0, no pad/iterate
  g1[1] = (int)((tensor_d0 & 0xffffu) << 16);   // [47:32]=0, tensor_d0[15:0]
  g1[2] = (int)((tensor_d0 >> 16) | ((tensor_d1 & 0xffffu) << 16));
  g1[3] = (int)((tensor_d1 >> 16) | (tile_d0 << 16));
  g1[4] = (int)(tile_d1 & 0xffffu);             // tile_dim1; tile_dim2=0
  g1[5] = (int)(stride0_units & 0xffffffffu);   // tensor_dim0_stride[31:0]
  g1[6] = (int)((stride0_units >> 32) & 0xffffu); // stride[47:32]; d1_stride=0
  g1[7] = 0;
  i32x4 z4 = {0, 0, 0, 0};                      // groups 2/3 unused (2D tile)
  i32x8 z8 = {0, 0, 0, 0, 0, 0, 0, 0};
  __builtin_amdgcn_tensor_load_to_lds(g0, g1, z4, z4, z8, 0);
}

// ---------------------------------------------------------------------------
// Kernel 0: W [1024x128] f32 -> Wt [128x1024] bf16 (transposed, k contiguous)
// ---------------------------------------------------------------------------
__global__ void wtrans_kernel(const float* __restrict__ W, bf16* __restrict__ Wt) {
  int idx = blockIdx.x * 256 + threadIdx.x;   // 0 .. 1024*128
  int k = idx >> 7;
  int d = idx & 127;
  Wt[(size_t)d * D_MODEL + k] = (bf16)W[idx];
}

// ---------------------------------------------------------------------------
// Kernel 1: QKV projection. One block = one 16-row tile of x/z, 8 waves each
// computing a 16-wide slice of the 128 output dims for q, k, v.
// Q stored row-major bf16 with 1/sqrt(128) folded in; K row-major bf16;
// V stored TRANSPOSED as Vt[128][4096] bf16 for the attention kernel.
// W-operand global loads are software-pipelined one k-chunk ahead.
// ---------------------------------------------------------------------------
__global__ void __launch_bounds__(256) qkv_kernel(
    const float* __restrict__ x, const float* __restrict__ z,
    const bf16* __restrict__ Wqt, const bf16* __restrict__ Wkt,
    const bf16* __restrict__ Wvt,
    const float* __restrict__ bqv, const float* __restrict__ bkv,
    const float* __restrict__ bvv,
    bf16* __restrict__ Q, bf16* __restrict__ Kb, bf16* __restrict__ Vt) {
  __shared__ __align__(16) bf16 xs[16 * 256];
  __shared__ __align__(16) bf16 zs[16 * 256];

  const int tid  = threadIdx.x;
  const int lane = tid & 31;
  const int wave = tid >> 5;          // 0..7
  const int ln16 = lane & 15;
  const int hi   = lane >> 4;         // lane half selects K sub-range
  const int row_base = blockIdx.x * 16;
  const int d_col = wave * 16 + ln16; // output column this lane owns (B operand)

  const bf16* wq_row = Wqt + (size_t)d_col * D_MODEL;
  const bf16* wk_row = Wkt + (size_t)d_col * D_MODEL;
  const bf16* wv_row = Wvt + (size_t)d_col * D_MODEL;

  v8f accQ = {0,0,0,0,0,0,0,0};
  v8f accK = {0,0,0,0,0,0,0,0};
  v8f accV = {0,0,0,0,0,0,0,0};

  for (int kc = 0; kc < D_MODEL; kc += 256) {
    __syncthreads();   // previous phase fully consumed
    // Stage x/z chunk [16 x 256] f32 -> bf16 in LDS (coalesced b128 loads).
    for (int i = tid; i < 1024; i += 256) {
      int row = i >> 6;
      int col = (i << 2) & 255;
      f32x4 fx = *reinterpret_cast<const f32x4*>(
          x + (size_t)(row_base + row) * D_MODEL + kc + col);
      f32x4 fz = *reinterpret_cast<const f32x4*>(
          z + (size_t)(row_base + row) * D_MODEL + kc + col);
      Pack4 px, pz;
      px.h[0] = (bf16)fx.x; px.h[1] = (bf16)fx.y;
      px.h[2] = (bf16)fx.z; px.h[3] = (bf16)fx.w;
      pz.h[0] = (bf16)fz.x; pz.h[1] = (bf16)fz.y;
      pz.h[2] = (bf16)fz.z; pz.h[3] = (bf16)fz.w;
      reinterpret_cast<u32x2*>(xs)[i] = px.u;
      reinterpret_cast<u32x2*>(zs)[i] = pz.u;
    }
    __syncthreads();

    const bf16* xrow = xs + ln16 * 256;
    const bf16* zrow = zs + ln16 * 256;

    // Preload W fragments for chunk 0 of this phase.
    ABFrag fq, fk, fv;
    {
      int g0 = kc + hi * 8, g1 = g0 + 16;
      fq.u[0] = *reinterpret_cast<const u32x4*>(wq_row + g0);
      fq.u[1] = *reinterpret_cast<const u32x4*>(wq_row + g1);
      fk.u[0] = *reinterpret_cast<const u32x4*>(wk_row + g0);
      fk.u[1] = *reinterpret_cast<const u32x4*>(wk_row + g1);
      fv.u[0] = *reinterpret_cast<const u32x4*>(wv_row + g0);
      fv.u[1] = *reinterpret_cast<const u32x4*>(wv_row + g1);
    }
#pragma unroll
    for (int c = 0; c < 8; ++c) {           // 8 k-chunks of 32 per phase
      int o0 = c * 32 + hi * 8;
      int o1 = o0 + 16;
      ABFrag ax, az;
      ax.u[0] = *reinterpret_cast<const u32x4*>(xrow + o0);
      ax.u[1] = *reinterpret_cast<const u32x4*>(xrow + o1);
      az.u[0] = *reinterpret_cast<const u32x4*>(zrow + o0);
      az.u[1] = *reinterpret_cast<const u32x4*>(zrow + o1);
      ABFrag nq, nk, nv;
      if (c < 7) {                          // prefetch next chunk's W frags
        int g0 = kc + o0 + 32, g1 = g0 + 16;
        nq.u[0] = *reinterpret_cast<const u32x4*>(wq_row + g0);
        nq.u[1] = *reinterpret_cast<const u32x4*>(wq_row + g1);
        nk.u[0] = *reinterpret_cast<const u32x4*>(wk_row + g0);
        nk.u[1] = *reinterpret_cast<const u32x4*>(wk_row + g1);
        nv.u[0] = *reinterpret_cast<const u32x4*>(wv_row + g0);
        nv.u[1] = *reinterpret_cast<const u32x4*>(wv_row + g1);
      }
      accQ = wmma_bf16(ax.v, fq.v, accQ);
      accK = wmma_bf16(az.v, fk.v, accK);
      accV = wmma_bf16(az.v, fv.v, accV);
      if (c < 7) { fq = nq; fk = nk; fv = nv; }
    }
  }

  // Epilogue. D layout: element j -> row m = j + 8*hi, col = d_col.
  const float inv_sqrt_d = 0.088388347648318447f;  // 1/sqrt(128)
  float biasq = bqv[d_col];
  float biask = bkv[d_col];
  float biasv = bvv[d_col];
  Pack8 vp;
#pragma unroll
  for (int j = 0; j < 8; ++j) {
    int mrow = row_base + j + 8 * hi;
    Q[(size_t)mrow * D_ATTN + d_col]  = (bf16)((accQ[j] + biasq) * inv_sqrt_d);
    Kb[(size_t)mrow * D_ATTN + d_col] = (bf16)(accK[j] + biask);
    vp.h[j] = (bf16)(accV[j] + biasv);
  }
  // V transposed: lane holds 8 consecutive source rows for fixed d_col -> b128.
  *reinterpret_cast<u32x4*>(Vt + (size_t)d_col * SEQ_L + row_base + 8 * hi) = vp.u;
}

// ---------------------------------------------------------------------------
// Kernel 2: flash attention. 2 waves/block (128 blocks), each wave owns 16
// query rows (Q held in registers). K/V^T chunks of 32 keys double-buffered
// in LDS, filled by the Tensor Data Mover (wave 0 issues, s_wait_tensorcnt).
// Computes S^T = K*Q^T (per-lane row stats), then y^T += V^T * P^T where the
// exp'd scores are already in B-operand layout (no transpose needed).
// ---------------------------------------------------------------------------
__global__ void __launch_bounds__(64) attn_kernel(
    const bf16* __restrict__ Q, const bf16* __restrict__ Kb,
    const bf16* __restrict__ Vt, float* __restrict__ y) {
  __shared__ __align__(16) bf16 Ks[2][32 * 128];   // 32 keys x 128 feats
  __shared__ __align__(16) bf16 Vs[2][128 * 32];   // 128 dims x 32 keys

  const int tid  = threadIdx.x;
  const int lane = tid & 31;
  const int wave = tid >> 5;          // 0..1
  const int ln16 = lane & 15;
  const int hi   = lane >> 4;
  const int qrow = blockIdx.x * 32 + wave * 16 + ln16;

  // Q rows in B-operand layout: 4 chunks over k = 128.
  ABFrag qb[4];
  const bf16* qrp = Q + (size_t)qrow * D_ATTN;
#pragma unroll
  for (int c = 0; c < 4; ++c) {
    int o0 = c * 32 + hi * 8;
    qb[c].u[0] = *reinterpret_cast<const u32x4*>(qrp + o0);
    qb[c].u[1] = *reinterpret_cast<const u32x4*>(qrp + o0 + 16);
  }

  v8f yacc[8];
#pragma unroll
  for (int t = 0; t < 8; ++t) yacc[t] = (v8f){0,0,0,0,0,0,0,0};
  float row_max = -3.0e38f;
  float row_sum = 0.0f;

  // TDM staging of one 32-key chunk into LDS buffer (it & 1).
  auto stage = [&](int it) {
    int buf = it & 1;
    size_t kb = (size_t)it * 32;
    unsigned ldsK = (unsigned)(unsigned long)(const void*)&Ks[buf][0];
    unsigned ldsV = (unsigned)(unsigned long)(const void*)&Vs[buf][0];
    // K chunk: contiguous 8KB -> 1D tile of 1024 x 8B.
    tdm_load_2d(ldsK, Kb + kb * D_ATTN, /*tile_d0=*/1024, /*tile_d1=*/0,
                /*stride0=*/1024);
    // V^T chunk: 128 rows x 64B, global row stride 8192B -> 2D tile 8x128.
    tdm_load_2d(ldsV, Vt + kb, /*tile_d0=*/8, /*tile_d1=*/128,
                /*stride0=*/1024);
  };

  if (wave == 0) stage(0);

  for (int it = 0; it < 128; ++it) {
    if (wave == 0) __builtin_amdgcn_s_wait_tensorcnt(0);  // chunk it in LDS
    __syncthreads();                 // chunk it visible; buf(it+1) free
    if (wave == 0 && it + 1 < 128) stage(it + 1);
    int buf = it & 1;

    // --- S^T = K_tile * Q^T : two 16x16 tiles (keys 0-15, 16-31 of chunk) ---
    const bf16* ka0 = Ks[buf] + ln16 * 128;
    const bf16* ka1 = Ks[buf] + (16 + ln16) * 128;
    v8f s0 = {0,0,0,0,0,0,0,0};
    v8f s1 = {0,0,0,0,0,0,0,0};
#pragma unroll
    for (int c = 0; c < 4; ++c) {
      int o0 = c * 32 + hi * 8;
      ABFrag a0, a1;
      a0.u[0] = *reinterpret_cast<const u32x4*>(ka0 + o0);
      a0.u[1] = *reinterpret_cast<const u32x4*>(ka0 + o0 + 16);
      a1.u[0] = *reinterpret_cast<const u32x4*>(ka1 + o0);
      a1.u[1] = *reinterpret_cast<const u32x4*>(ka1 + o0 + 16);
      s0 = wmma_bf16(a0.v, qb[c].v, s0);
      s1 = wmma_bf16(a1.v, qb[c].v, s1);
    }
    // Lane holds: s0[j] = S[q=ln16][key = j+8*hi], s1[j] = same +16 keys.

    // --- online softmax (per-query stats live in lanes q and q+16) ---
    float mx = s0[0];
#pragma unroll
    for (int j = 1; j < 8; ++j) mx = fmaxf(mx, s0[j]);
#pragma unroll
    for (int j = 0; j < 8; ++j) mx = fmaxf(mx, s1[j]);
    mx = fmaxf(mx, __shfl_xor(mx, 16, 32));
    float nmax = fmaxf(row_max, mx);
    float corr = __expf(row_max - nmax);
    row_max = nmax;

    float p0[8], p1[8];
    float psum = 0.0f;
#pragma unroll
    for (int j = 0; j < 8; ++j) {
      p0[j] = __expf(s0[j] - nmax);
      p1[j] = __expf(s1[j] - nmax);
      psum += p0[j] + p1[j];
    }
    psum += __shfl_xor(psum, 16, 32);
    row_sum = row_sum * corr + psum;

#pragma unroll
    for (int t = 0; t < 8; ++t)
#pragma unroll
      for (int j = 0; j < 8; ++j) yacc[t][j] *= corr;

    // P^T already matches the B-operand layout: element e needs
    // key = (e>>3)*16 + hi*8 + (e&7)  ->  p0[e] / p1[e-8].
    ABFrag pb;
#pragma unroll
    for (int e = 0; e < 8; ++e) {
      pb.v[e]     = (bf16)p0[e];
      pb.v[e + 8] = (bf16)p1[e];
    }

    // --- y^T += V^T_tile * P^T, 8 tiles over the 128 value dims ---
#pragma unroll
    for (int t = 0; t < 8; ++t) {
      const bf16* vr = Vs[buf] + (t * 16 + ln16) * 32;
      ABFrag va;
      va.u[0] = *reinterpret_cast<const u32x4*>(vr + hi * 8);
      va.u[1] = *reinterpret_cast<const u32x4*>(vr + 16 + hi * 8);
      yacc[t] = wmma_bf16(va.v, pb.v, yacc[t]);
    }
  }

  // --- epilogue: y[q][d] = yacc / row_sum ; d = 16t + 8*hi + j contiguous ---
  float inv = 1.0f / row_sum;
  float* yrow = y + (size_t)qrow * D_ATTN;
#pragma unroll
  for (int t = 0; t < 8; ++t) {
    f32x4 a, b;
#pragma unroll
    for (int j = 0; j < 4; ++j) {
      a[j] = yacc[t][j] * inv;
      b[j] = yacc[t][j + 4] * inv;
    }
    int dcol = t * 16 + hi * 8;
    *reinterpret_cast<f32x4*>(yrow + dcol)     = a;
    *reinterpret_cast<f32x4*>(yrow + dcol + 4) = b;
  }
}

// ---------------------------------------------------------------------------
// Host launch.
// Workspace layout (bytes):
//   [0,       256K) Wqt bf16 [128][1024]
//   [256K,    512K) Wkt
//   [512K,    768K) Wvt
//   [768K,   1792K) Q  bf16 [4096][128]
//   [1792K,  2816K) K  bf16 [4096][128]
//   [2816K,  3840K) Vt bf16 [128][4096]
// ---------------------------------------------------------------------------
extern "C" void kernel_launch(void* const* d_in, const int* in_sizes, int n_in,
                              void* d_out, int out_size, void* d_ws, size_t ws_size,
                              hipStream_t stream) {
  (void)in_sizes; (void)n_in; (void)out_size; (void)ws_size;
  const float* x  = (const float*)d_in[0];
  const float* z  = (const float*)d_in[1];
  const float* Wq = (const float*)d_in[2];
  const float* bq = (const float*)d_in[3];
  const float* Wk = (const float*)d_in[4];
  const float* bk = (const float*)d_in[5];
  const float* Wv = (const float*)d_in[6];
  const float* bv = (const float*)d_in[7];

  char* ws = (char*)d_ws;
  bf16* Wqt = (bf16*)(ws);
  bf16* Wkt = (bf16*)(ws + (256u << 10));
  bf16* Wvt = (bf16*)(ws + (512u << 10));
  bf16* Q   = (bf16*)(ws + (768u << 10));
  bf16* Kb  = (bf16*)(ws + (768u << 10) + (1024u << 10));
  bf16* Vt  = (bf16*)(ws + (768u << 10) + (2048u << 10));

  wtrans_kernel<<<512, 256, 0, stream>>>(Wq, Wqt);
  wtrans_kernel<<<512, 256, 0, stream>>>(Wk, Wkt);
  wtrans_kernel<<<512, 256, 0, stream>>>(Wv, Wvt);

  qkv_kernel<<<SEQ_L / 16, 256, 0, stream>>>(x, z, Wqt, Wkt, Wvt,
                                             bq, bk, bv, Q, Kb, Vt);

  attn_kernel<<<SEQ_L / 32, 64, 0, stream>>>(Q, Kb, Vt, (float*)d_out);
}
